// MultiHeadLatentAttention_32152125178320
// MI455X (gfx1250) — matrix operand order repaired and checked by reading, not compiled, not run on hardware
//
#include <hip/hip_runtime.h>
#include <hip/hip_bf16.h>
#include <math.h>

typedef __bf16 bf16;
typedef __attribute__((ext_vector_type(8)))  __bf16 bf16x8;
typedef __attribute__((ext_vector_type(16))) __bf16 bf16x16;
typedef __attribute__((ext_vector_type(8)))  float  v8f;

#define B_    2
#define S_    2048
#define DIM_  2048
#define H_    16
#define NOPE_ 128
#define ROPE_ 64
#define VD_   128
#define QKD_  192   // NOPE + ROPE
#define QR_   1024
#define KVR_  512
#define SCALE_ 0.07216878364870323f  // 1/sqrt(192), folded into q at projection time

enum { EP_F32 = 0, EP_BF16 = 1, EP_QNOPE = 2, EP_QPE = 3, EP_KV = 4 };

static __device__ __forceinline__ bf16x16 cat8(bf16x8 a, bf16x8 b) {
  return __builtin_shufflevector(a, b, 0,1,2,3,4,5,6,7,8,9,10,11,12,13,14,15);
}
static __device__ __forceinline__ v8f wmma_bf16(bf16x16 a, bf16x16 b, v8f c) {
  return __builtin_amdgcn_wmma_f32_16x16x32_bf16(false, a, false, b, (short)0, c, false, false);
}
// A fragment 16x32 (MxK) at (m0,k0) from row-major [M,lda] in GLOBAL memory.
static __device__ __forceinline__ bf16x16 loadA(const bf16* A, int lda, int m0, int k0,
                                                int l16, int half) {
  const bf16* p = A + (size_t)(m0 + l16) * lda + k0 + half * 8;
  return cat8(*(const bf16x8*)p, *(const bf16x8*)(p + 16));
}
// B fragment 32x16 (KxN): lane = column n (row of row-major W[N,K]), two 16B chunks at k0+half*8 and +16.
static __device__ __forceinline__ bf16x16 loadB(const bf16* W, int ldb, int n0, int k0,
                                                int l16, int half) {
  const bf16* p = W + (size_t)(n0 + l16) * ldb + k0 + half * 8;
  return cat8(*(const bf16x8*)p, *(const bf16x8*)(p + 16));
}
// Async global->LDS copy of one 16B chunk per lane (CDNA5 async path, ASYNCcnt).
static __device__ __forceinline__ void async_cp16(const void* gptr, void* lptr) {
  uint32_t lds = (uint32_t)(uintptr_t)lptr;          // low 32 bits = LDS byte offset
  uint64_t ga  = (uint64_t)(uintptr_t)gptr;
  asm volatile("global_load_async_to_lds_b128 %0, %1, off" :: "v"(lds), "v"(ga) : "memory");
}
static __device__ __forceinline__ void wait_async0() {
  asm volatile("s_wait_asynccnt 0x0" ::: "memory");
}

// ---------------------------------------------------------------- cast
__global__ void cast_f32_bf16(const float* __restrict__ in, bf16* __restrict__ out, size_t n) {
  size_t i = (size_t)blockIdx.x * blockDim.x + threadIdx.x;
  size_t st = (size_t)gridDim.x * blockDim.x;
  for (; i < n; i += st) out[i] = (bf16)in[i];
}

// ---------------------------------------------------------------- rmsnorm (f32 in, bf16 out)
__global__ __launch_bounds__(256) void rmsnorm_bf16(const float* __restrict__ in,
                                                    const float* __restrict__ w,
                                                    bf16* __restrict__ out, int width) {
  int row = blockIdx.x, tid = threadIdx.x;
  const float* r = in + (size_t)row * width;
  float ss = 0.f;
  for (int i = tid; i < width; i += 256) { float v = r[i]; ss += v * v; }
  __shared__ float red[256];
  red[tid] = ss; __syncthreads();
  for (int s = 128; s > 0; s >>= 1) { if (tid < s) red[tid] += red[tid + s]; __syncthreads(); }
  float inv = rsqrtf(red[0] / (float)width + 1e-6f);
  bf16* o = out + (size_t)row * width;
  for (int i = tid; i < width; i += 256) o[i] = (bf16)(r[i] * inv * w[i]);
}

// ---------------------------------------------------------------- generic WMMA GEMM
// C[M,N] = A[M,K] * W[N,K]^T. Block = 8 waves; wave computes 64x32, block tile 256x64.
// A/B tiles double-buffered in LDS via GLOBAL_LOAD_ASYNC_TO_LDS_B128 (overlaps the
// k+1 copy with WMMA on k). LDS rows padded to 40 elems (80B) -> 16-lane ds_load_b128
// hits 16 distinct banks (20-bank stride, gcd 4 with 64).
#define LDA_P 40
template<int MODE>
__global__ __launch_bounds__(256) void gemm_bf16_wmma(const bf16* __restrict__ A,
                                                      const bf16* __restrict__ W,
                                                      void* __restrict__ C0v,
                                                      void* __restrict__ C1v,
                                                      int K, int lda, int ldb, int ldc) {
  __shared__ __attribute__((aligned(16))) bf16 Asm[2][256 * LDA_P];
  __shared__ __attribute__((aligned(16))) bf16 Bsm[2][64 * LDA_P];
  const int tid  = threadIdx.x;
  const int lane = tid & 31;
  const int half = lane >> 4, l16 = lane & 15;
  const int wid  = tid >> 5;
  const int wm = wid & 3, wn = wid >> 2;
  const int mblk = blockIdx.y * 256;
  const int nblk = blockIdx.x * 64;

  v8f acc[4][2];
#pragma unroll
  for (int mt = 0; mt < 4; ++mt)
#pragma unroll
    for (int nt = 0; nt < 2; ++nt) acc[mt][nt] = (v8f)0.f;

  auto stage = [&](int buf, int k0) {
    // A tile: 256 rows x 32 cols, 4 chunks/row -> 1024 chunks, 4 per thread
#pragma unroll
    for (int c = 0; c < 4; ++c) {
      int chunk = tid + c * 256;
      int row = chunk >> 2, col = (chunk & 3) * 8;
      async_cp16(A + (size_t)(mblk + row) * lda + k0 + col, &Asm[buf][row * LDA_P + col]);
    }
    // B tile: 64 rows x 32 cols -> 256 chunks, 1 per thread
    {
      int row = tid >> 2, col = (tid & 3) * 8;
      async_cp16(W + (size_t)(nblk + row) * ldb + k0 + col, &Bsm[buf][row * LDA_P + col]);
    }
  };

  stage(0, 0);
  wait_async0();
  __syncthreads();

  for (int k0 = 0; k0 < K; k0 += 32) {
    const int cur = (k0 >> 5) & 1;
    const bool more = (k0 + 32) < K;
    if (more) stage(cur ^ 1, k0 + 32);

    bf16x16 af[4], bfg[2];
#pragma unroll
    for (int mt = 0; mt < 4; ++mt) {
      const bf16* p = &Asm[cur][(wm * 64 + mt * 16 + l16) * LDA_P + half * 8];
      af[mt] = cat8(*(const bf16x8*)p, *(const bf16x8*)(p + 16));
    }
#pragma unroll
    for (int nt = 0; nt < 2; ++nt) {
      const bf16* p = &Bsm[cur][(wn * 32 + nt * 16 + l16) * LDA_P + half * 8];
      bfg[nt] = cat8(*(const bf16x8*)p, *(const bf16x8*)(p + 16));
    }
#pragma unroll
    for (int mt = 0; mt < 4; ++mt)
#pragma unroll
      for (int nt = 0; nt < 2; ++nt) acc[mt][nt] = wmma_bf16(af[mt], bfg[nt], acc[mt][nt]);

    if (more) wait_async0();
    __syncthreads();
  }

#pragma unroll
  for (int mt = 0; mt < 4; ++mt)
#pragma unroll
    for (int nt = 0; nt < 2; ++nt)
#pragma unroll
      for (int r = 0; r < 8; ++r) {
        int m = mblk + wm * 64 + mt * 16 + half * 8 + r;
        int n = nblk + wn * 32 + nt * 16 + l16;
        float v = acc[mt][nt][r];
        if constexpr (MODE == EP_F32) {
          ((float*)C0v)[(size_t)m * ldc + n] = v;
        } else if constexpr (MODE == EP_BF16) {
          ((bf16*)C0v)[(size_t)m * ldc + n] = (bf16)v;
        } else if constexpr (MODE == EP_QNOPE) {  // pre-scaled by 1/sqrt(192)
          int b = m >> 11, s = m & 2047, h = n >> 7, d = n & 127;
          ((bf16*)C0v)[((size_t)(b * H_ + h) * S_ + s) * QKD_ + d] = (bf16)(v * SCALE_);
        } else if constexpr (MODE == EP_QPE) {    // pre-scaled; RoPE rotation commutes
          int b = m >> 11, s = m & 2047, h = n >> 6, d = n & 63;
          ((bf16*)C0v)[((size_t)(b * H_ + h) * S_ + s) * QKD_ + NOPE_ + d] = (bf16)(v * SCALE_);
        } else {  // EP_KV: n = h*256 + j ; j<128 -> k_nope into k_full ; else v into vT[B,H,VD,S]
          int b = m >> 11, s = m & 2047, h = n >> 8, j = n & 255;
          if (j < NOPE_)
            ((bf16*)C0v)[((size_t)(b * H_ + h) * S_ + s) * QKD_ + j] = (bf16)v;
          else
            ((bf16*)C1v)[((size_t)(b * H_ + h) * VD_ + (j - NOPE_)) * S_ + s] = (bf16)v;
        }
      }
  (void)C1v; (void)ldc;
}

// ---------------------------------------------------------------- RoPE (interleaved pairs)
__global__ __launch_bounds__(256) void rope_kernel(bf16* __restrict__ q_full,
                                                   bf16* __restrict__ k_full,
                                                   const bf16* __restrict__ krope) {
  int bs = blockIdx.x, b = bs >> 11, s = bs & 2047, tid = threadIdx.x;
  const float LOGC = 0.28782313662425572f;  // ln(10000)/32
  for (int idx = tid; idx < H_ * 32; idx += 256) {
    int h = idx >> 5, i = idx & 31;
    float ang = (float)s * __expf(-(float)i * LOGC);
    float c = __cosf(ang), sn = __sinf(ang);
    size_t base = ((size_t)(b * H_ + h) * S_ + s) * QKD_ + NOPE_ + 2 * i;
    float x1 = (float)q_full[base], x2 = (float)q_full[base + 1];
    q_full[base]     = (bf16)(x1 * c - x2 * sn);
    q_full[base + 1] = (bf16)(x1 * sn + x2 * c);
  }
  if (tid < 32) {
    int i = tid;
    float ang = (float)s * __expf(-(float)i * LOGC);
    float c = __cosf(ang), sn = __sinf(ang);
    size_t kb = ((size_t)(b * S_) + s) * ROPE_ + 2 * i;
    float x1 = (float)krope[kb], x2 = (float)krope[kb + 1];
    float o1 = x1 * c - x2 * sn, o2 = x1 * sn + x2 * c;
    for (int h = 0; h < H_; ++h) {
      size_t base = ((size_t)(b * H_ + h) * S_ + s) * QKD_ + NOPE_ + 2 * i;
      k_full[base] = (bf16)o1; k_full[base + 1] = (bf16)o2;
    }
  }
}

// ---------------------------------------------------------------- flash attention
// Wave handles 16 queries of one (b,h); 8 waves/block -> 128 queries/block.
// Per 32-key block: 12 WMMA (QK^T over d=192) + online softmax + 8 WMMA (PV).
// Q pre-scaled by 1/sqrt(192) at projection time.
__global__ __launch_bounds__(256) void mla_flash(const bf16* __restrict__ q_full,
                                                 const bf16* __restrict__ k_full,
                                                 const bf16* __restrict__ vT,
                                                 bf16* __restrict__ attn) {
  const int b = blockIdx.z, h = blockIdx.y;
  const int wid  = threadIdx.x >> 5;
  const int lane = threadIdx.x & 31;
  const int half = lane >> 4, l16 = lane & 15;
  const int q0 = blockIdx.x * 128 + wid * 16;

  const bf16* Qb = q_full + (size_t)(b * H_ + h) * S_ * QKD_;
  const bf16* Kb = k_full + (size_t)(b * H_ + h) * S_ * QKD_;
  const bf16* Vt = vT     + (size_t)(b * H_ + h) * VD_ * S_;

  __shared__ __attribute__((aligned(16))) bf16 Pl[8][16 * 32];

  bf16x16 qf[6];
#pragma unroll
  for (int t = 0; t < 6; ++t) qf[t] = loadA(Qb, QKD_, q0, 32 * t, l16, half);

  v8f o[8];
#pragma unroll
  for (int j = 0; j < 8; ++j) o[j] = (v8f)0.f;
  float mrow[8], lrow[8];
#pragma unroll
  for (int r = 0; r < 8; ++r) { mrow[r] = -INFINITY; lrow[r] = 0.f; }

  const int kend = q0 + 16;  // causal: keys <= q0+15
  for (int kb = 0; kb < kend; kb += 32) {
    v8f s0 = (v8f)0.f, s1 = (v8f)0.f;
#pragma unroll
    for (int t = 0; t < 6; ++t) {
      bf16x16 b0 = loadB(Kb, QKD_, kb,      32 * t, l16, half);
      bf16x16 b1 = loadB(Kb, QKD_, kb + 16, 32 * t, l16, half);
      s0 = wmma_bf16(qf[t], b0, s0);
      s1 = wmma_bf16(qf[t], b1, s1);
    }
#pragma unroll
    for (int r = 0; r < 8; ++r) {
      int qm = q0 + half * 8 + r;
      float e0 = (kb + l16      <= qm) ? s0[r] : -INFINITY;
      float e1 = (kb + 16 + l16 <= qm) ? s1[r] : -INFINITY;
      float mx = fmaxf(e0, e1);
#pragma unroll
      for (int off = 8; off >= 1; off >>= 1) mx = fmaxf(mx, __shfl_xor(mx, off, 16));
      float mnew  = fmaxf(mrow[r], mx);
      float alpha = __expf(mrow[r] - mnew);
      float p0 = __expf(e0 - mnew), p1 = __expf(e1 - mnew);
      float rs = p0 + p1;
#pragma unroll
      for (int off = 8; off >= 1; off >>= 1) rs += __shfl_xor(rs, off, 16);
      lrow[r] = lrow[r] * alpha + rs;
      mrow[r] = mnew;
#pragma unroll
      for (int j = 0; j < 8; ++j) o[j][r] *= alpha;
      int rr = half * 8 + r;
      Pl[wid][rr * 32 + l16]      = (bf16)p0;
      Pl[wid][rr * 32 + 16 + l16] = (bf16)p1;
    }
    // P tile (16x32) C-layout -> A-fragment via per-wave LDS
    const bf16* pp = &Pl[wid][(size_t)l16 * 32 + half * 8];
    bf16x16 pf = cat8(*(const bf16x8*)pp, *(const bf16x8*)(pp + 16));
#pragma unroll
    for (int j = 0; j < 8; ++j) {
      const bf16* vp = Vt + (size_t)(j * 16 + l16) * S_ + kb + half * 8;
      bf16x16 bv = cat8(*(const bf16x8*)vp, *(const bf16x8*)(vp + 16));
      o[j] = wmma_bf16(pf, bv, o[j]);
    }
  }
#pragma unroll
  for (int j = 0; j < 8; ++j)
#pragma unroll
    for (int r = 0; r < 8; ++r) {
      int s = q0 + half * 8 + r;
      size_t dst = ((size_t)(b * S_ + s)) * (H_ * VD_) + h * VD_ + j * 16 + l16;
      attn[dst] = (bf16)(o[j][r] / lrow[r]);
    }
}

// ---------------------------------------------------------------- launch
extern "C" void kernel_launch(void* const* d_in, const int* in_sizes, int n_in,
                              void* d_out, int out_size, void* d_ws, size_t ws_size,
                              hipStream_t stream) {
  (void)in_sizes; (void)n_in; (void)out_size; (void)ws_size;
  const float* x        = (const float*)d_in[0];
  const float* wq_down  = (const float*)d_in[1];
  const float* q_norm_w = (const float*)d_in[2];
  const float* wq_up    = (const float*)d_in[3];
  const float* wq_rope  = (const float*)d_in[4];
  const float* wkv_down = (const float*)d_in[5];
  const float* kv_norm_w= (const float*)d_in[6];
  const float* wkv_up   = (const float*)d_in[7];
  const float* wk_rope  = (const float*)d_in[8];
  const float* wo       = (const float*)d_in[9];
  float* out = (float*)d_out;

  char* ws = (char*)d_ws;
  size_t off = 0;
  auto alloc = [&](size_t bytes) -> char* {
    char* p = ws + off; off += (bytes + 255) & ~(size_t)255; return p;
  };
  const size_t M = (size_t)B_ * S_;  // 4096
  bf16* xb    = (bf16*)alloc(M * DIM_ * 2);
  bf16* wqd   = (bf16*)alloc((size_t)QR_ * DIM_ * 2);
  bf16* wqu   = (bf16*)alloc((size_t)H_ * NOPE_ * QR_ * 2);
  bf16* wqr   = (bf16*)alloc((size_t)H_ * ROPE_ * QR_ * 2);
  bf16* wkd   = (bf16*)alloc((size_t)KVR_ * DIM_ * 2);
  bf16* wku   = (bf16*)alloc((size_t)H_ * (NOPE_ + VD_) * KVR_ * 2);
  bf16* wkr   = (bf16*)alloc((size_t)ROPE_ * DIM_ * 2);
  bf16* wob   = (bf16*)alloc((size_t)DIM_ * H_ * VD_ * 2);
  float* tmpf = (float*)alloc(M * QR_ * 4);
  bf16* qcb   = (bf16*)alloc(M * QR_ * 2);
  bf16* kvcb  = (bf16*)alloc(M * KVR_ * 2);
  bf16* qfull = (bf16*)alloc((size_t)B_ * H_ * S_ * QKD_ * 2);
  bf16* kfull = (bf16*)alloc((size_t)B_ * H_ * S_ * QKD_ * 2);
  bf16* vT    = (bf16*)alloc((size_t)B_ * H_ * VD_ * S_ * 2);
  bf16* krope = (bf16*)alloc(M * ROPE_ * 2);
  bf16* attn  = (bf16*)alloc(M * H_ * VD_ * 2);

  auto cast = [&](const float* src, bf16* dst, size_t n) {
    cast_f32_bf16<<<2048, 256, 0, stream>>>(src, dst, n);
  };
  cast(x,        xb,  M * DIM_);
  cast(wq_down,  wqd, (size_t)QR_ * DIM_);
  cast(wq_up,    wqu, (size_t)H_ * NOPE_ * QR_);
  cast(wq_rope,  wqr, (size_t)H_ * ROPE_ * QR_);
  cast(wkv_down, wkd, (size_t)KVR_ * DIM_);
  cast(wkv_up,   wku, (size_t)H_ * (NOPE_ + VD_) * KVR_);
  cast(wk_rope,  wkr, (size_t)ROPE_ * DIM_);
  cast(wo,       wob, (size_t)DIM_ * H_ * VD_);

  dim3 blk(256);
  const int MB = (int)(M / 256);  // 16 M-blocks of 256
  // q_c = rmsnorm(x @ wq_down^T)
  gemm_bf16_wmma<EP_F32><<<dim3(QR_ / 64, MB), blk, 0, stream>>>(xb, wqd, tmpf, nullptr, DIM_, DIM_, DIM_, QR_);
  rmsnorm_bf16<<<(int)M, 256, 0, stream>>>(tmpf, q_norm_w, qcb, QR_);
  // kv_c = rmsnorm(x @ wkv_down^T)
  gemm_bf16_wmma<EP_F32><<<dim3(KVR_ / 64, MB), blk, 0, stream>>>(xb, wkd, tmpf, nullptr, DIM_, DIM_, DIM_, KVR_);
  rmsnorm_bf16<<<(int)M, 256, 0, stream>>>(tmpf, kv_norm_w, kvcb, KVR_);
  // q_nope / q_pe scattered into q_full[B,H,S,192] (pre-scaled)
  gemm_bf16_wmma<EP_QNOPE><<<dim3(H_ * NOPE_ / 64, MB), blk, 0, stream>>>(qcb, wqu, qfull, nullptr, QR_, QR_, QR_, 0);
  gemm_bf16_wmma<EP_QPE><<<dim3(H_ * ROPE_ / 64, MB), blk, 0, stream>>>(qcb, wqr, qfull, nullptr, QR_, QR_, QR_, 0);
  // kv_up scattered into k_full (nope part) + vT[B,H,VD,S]
  gemm_bf16_wmma<EP_KV><<<dim3(H_ * (NOPE_ + VD_) / 64, MB), blk, 0, stream>>>(kvcb, wku, kfull, vT, KVR_, KVR_, KVR_, 0);
  // shared k_rope (pre-rope)
  gemm_bf16_wmma<EP_BF16><<<dim3(ROPE_ / 64, MB), blk, 0, stream>>>(xb, wkr, krope, nullptr, DIM_, DIM_, DIM_, ROPE_);
  // rope on q_full pe-slice in place; rope+broadcast k_rope into k_full
  rope_kernel<<<(int)M, 256, 0, stream>>>(qfull, kfull, krope);
  // flash attention -> attn[B,S,H*VD] bf16
  mla_flash<<<dim3(S_ / 128, H_, B_), 256, 0, stream>>>(qfull, kfull, vT, attn);
  // final projection -> f32 out
  gemm_bf16_wmma<EP_F32><<<dim3(DIM_ / 64, MB), blk, 0, stream>>>(attn, wob, out, nullptr, H_ * VD_, H_ * VD_, H_ * VD_, DIM_);
}
